// TopKSAE_6081673691200
// MI455X (gfx1250) — matrix operand order repaired and checked by reading, not compiled, not run on hardware
//
#include <hip/hip_runtime.h>
#include <stdint.h>

#define B_ROWS 16384
#define D_DIM  1024
#define H_DIM  4096
#define K_TOP  16

typedef __bf16 bf16_t;
typedef __attribute__((ext_vector_type(16))) __bf16 v16bf;
typedef __attribute__((ext_vector_type(8)))  __bf16 v8bf;
typedef __attribute__((ext_vector_type(8)))  float  v8f;

union FragAB { v16bf f; v8bf h[2]; };

// ---------------------------------------------------------------- convert f32 -> bf16 (8 elems/thread)
__global__ __launch_bounds__(256)
void cvt_f32_bf16(const float* __restrict__ in, bf16_t* __restrict__ out, int n) {
  int i = (blockIdx.x * blockDim.x + threadIdx.x) * 8;
  if (i >= n) return;
  float4 a = *(const float4*)(in + i);
  float4 b = *(const float4*)(in + i + 4);
  union { unsigned short u[8]; uint4 q; } p;
  p.u[0] = __builtin_bit_cast(unsigned short, (bf16_t)a.x);
  p.u[1] = __builtin_bit_cast(unsigned short, (bf16_t)a.y);
  p.u[2] = __builtin_bit_cast(unsigned short, (bf16_t)a.z);
  p.u[3] = __builtin_bit_cast(unsigned short, (bf16_t)a.w);
  p.u[4] = __builtin_bit_cast(unsigned short, (bf16_t)b.x);
  p.u[5] = __builtin_bit_cast(unsigned short, (bf16_t)b.y);
  p.u[6] = __builtin_bit_cast(unsigned short, (bf16_t)b.z);
  p.u[7] = __builtin_bit_cast(unsigned short, (bf16_t)b.w);
  *(uint4*)(out + i) = p.q;
}

// ---------------------------------------------------------------- W_dec [D,H] -> Wt [H,D] tiled transpose
__global__ __launch_bounds__(256)
void transpose_wdec(const float* __restrict__ Wd, float* __restrict__ Wt) {
  __shared__ float tile[32][33];
  int x  = blockIdx.x * 32 + threadIdx.x;   // H index
  int y0 = blockIdx.y * 32;                 // D index base
  #pragma unroll
  for (int j = 0; j < 32; j += 8)
    tile[threadIdx.y + j][threadIdx.x] = Wd[(size_t)(y0 + threadIdx.y + j) * H_DIM + x];
  __syncthreads();
  int xo = blockIdx.x * 32;
  #pragma unroll
  for (int j = 0; j < 32; j += 8)
    Wt[(size_t)(xo + threadIdx.y + j) * D_DIM + y0 + threadIdx.x] = tile[threadIdx.x][threadIdx.y + j];
}

// ---------------------------------------------------------------- encoder GEMM: h = relu(x @ W_enc^T + b_enc)
// block tile 128(M) x 128(N), BK=64; 8 waves in 2(M) x 4(N) grid,
// each wave: 64x32 = 4x2 WMMA tiles -> 8 WMMAs per 12 ds_load_b128 (A-frags reused)
#define BM 128
#define BN 128
#define BK 64
#define LDA 72   /* BK + 8 bf16 pad: 144B row stride, conflict-free frag reads, 16B aligned */
#define LDB 72

__global__ __launch_bounds__(256)
void sae_encoder_gemm(const bf16_t* __restrict__ xb,
                      const bf16_t* __restrict__ wb,
                      const float*  __restrict__ b_enc,
                      float*        __restrict__ h) {
  __shared__ __align__(16) bf16_t As[BM * LDA];
  __shared__ __align__(16) bf16_t Bs[BN * LDB];

  const int tid  = threadIdx.x;
  const int lane = tid & 31;
  const int wave = tid >> 5;
  const int wm   = wave & 1;     // wave M slot (0..1) -> 64 rows each
  const int wn   = wave >> 1;    // wave N slot (0..3) -> 32 cols each
  const int hlf  = lane >> 4;    // lane half (ISA §7.12.2 layouts)
  const int l16  = lane & 15;

  const int blockM = blockIdx.x * BM;
  const int blockN = blockIdx.y * BN;

  v8f acc[4][2] = {};

  for (int k0 = 0; k0 < D_DIM; k0 += BK) {
    __syncthreads();
    // A tile: 128x64 bf16 = 1024 x 16B chunks, 4/thread
    #pragma unroll
    for (int t = 0; t < 4; ++t) {
      int c = tid + t * 256;
      int row = c >> 3, sub = c & 7;
      uint4 v = *(const uint4*)(xb + (size_t)(blockM + row) * D_DIM + k0 + sub * 8);
      *(uint4*)(&As[row * LDA + sub * 8]) = v;
    }
    // B tile: 128x64 bf16 = 1024 x 16B chunks, 4/thread
    #pragma unroll
    for (int t = 0; t < 4; ++t) {
      int c = tid + t * 256;
      int row = c >> 3, sub = c & 7;
      uint4 v = *(const uint4*)(wb + (size_t)(blockN + row) * D_DIM + k0 + sub * 8);
      *(uint4*)(&Bs[row * LDB + sub * 8]) = v;
    }
    __syncthreads();

    #pragma unroll
    for (int kk = 0; kk < 2; ++kk) {
      FragAB a[4], b[2];
      #pragma unroll
      for (int tm = 0; tm < 4; ++tm) {
        // A 16x32: lanes 0-15 -> K{0..7,16..23}, lanes 16-31 -> K{8..15,24..31}, row M = l16
        const bf16_t* p = &As[(wm * 64 + tm * 16 + l16) * LDA + kk * 32];
        a[tm].h[0] = *(const v8bf*)(p + hlf * 8);
        a[tm].h[1] = *(const v8bf*)(p + 16 + hlf * 8);
      }
      #pragma unroll
      for (int tn = 0; tn < 2; ++tn) {
        // B 32x16: lanes 0-15 -> K{0..7,16..23} of col N=l16, lanes 16-31 -> K{8..15,24..31}
        const bf16_t* p = &Bs[(wn * 32 + tn * 16 + l16) * LDB + kk * 32 + hlf * 8];
        b[tn].h[0] = *(const v8bf*)(p);
        b[tn].h[1] = *(const v8bf*)(p + 16);
      }
      #pragma unroll
      for (int tm = 0; tm < 4; ++tm)
        #pragma unroll
        for (int tn = 0; tn < 2; ++tn)
          acc[tm][tn] = __builtin_amdgcn_wmma_f32_16x16x32_bf16(
              false, a[tm].f, false, b[tn].f, (short)0, acc[tm][tn], false, false);
    }
  }

  // epilogue: + b_enc, ReLU, store dense h. D layout: VGPR i -> row i + 8*hlf, col = l16.
  #pragma unroll
  for (int tm = 0; tm < 4; ++tm) {
    #pragma unroll
    for (int tn = 0; tn < 2; ++tn) {
      int n = blockN + wn * 32 + tn * 16 + l16;
      float be = b_enc[n];
      int mbase = blockM + wm * 64 + tm * 16 + hlf * 8;
      #pragma unroll
      for (int i = 0; i < 8; ++i) {
        float r = fmaxf(acc[tm][tn][i] + be, 0.0f);
        h[(size_t)(mbase + i) * H_DIM + n] = r;
      }
    }
  }
}

// ---------------------------------------------------------------- top-16 per row, in-place mask + compact list
__global__ __launch_bounds__(256)
void sae_topk(float* __restrict__ h, float* __restrict__ tv, int* __restrict__ tix) {
  __shared__ float row[H_DIM];
  __shared__ unsigned long long red[256];
  const int tid = threadIdx.x;
  const size_t base = (size_t)blockIdx.x * H_DIM;

  float v[16];
  #pragma unroll
  for (int i = 0; i < 16; ++i) {
    int idx = tid + i * 256;
    v[i] = h[base + idx];
    row[idx] = v[i];
  }
  __syncthreads();

  for (int s = 0; s < K_TOP; ++s) {
    unsigned long long best = 0ull;
    #pragma unroll
    for (int i = 0; i < 16; ++i) {
      int idx = tid + i * 256;
      float val = row[idx];
      if (val >= 0.0f) {  // relu output; -1 marks already-selected
        unsigned long long key = ((unsigned long long)__float_as_uint(val) << 32)
                               | (unsigned)(H_DIM - 1 - idx);  // tie -> lower index wins
        best = best > key ? best : key;
      }
    }
    red[tid] = best;
    __syncthreads();
    for (int off = 128; off > 0; off >>= 1) {
      if (tid < off) {
        unsigned long long o = red[tid + off];
        if (o > red[tid]) red[tid] = o;
      }
      __syncthreads();
    }
    unsigned long long win = red[0];
    int widx = (H_DIM - 1) - (int)(win & 0xFFFFFFFFull);
    if (tid == (widx & 255)) row[widx] = -1.0f;   // owner marks selected
    if (tid == 0) {
      tv[blockIdx.x * K_TOP + s]  = __uint_as_float((unsigned)(win >> 32));
      tix[blockIdx.x * K_TOP + s] = widx;
    }
    __syncthreads();
  }

  #pragma unroll
  for (int i = 0; i < 16; ++i) {
    int idx = tid + i * 256;
    h[base + idx] = (row[idx] < 0.0f) ? v[i] : 0.0f;
  }
}

// ---------------------------------------------------------------- sparse decoder: out = h_sparse @ W_dec^T + b_dec
__global__ __launch_bounds__(256)
void sae_decoder(const float* __restrict__ tv, const int* __restrict__ tix,
                 const float* __restrict__ Wt, const float* __restrict__ b_dec,
                 float* __restrict__ out) {
  __shared__ float sv[K_TOP];
  __shared__ int   si[K_TOP];
  const int tid = threadIdx.x;
  const int b   = blockIdx.x;
  if (tid < K_TOP) {
    sv[tid] = tv[b * K_TOP + tid];
    si[tid] = tix[b * K_TOP + tid];
  }
  __syncthreads();
  int d0 = tid * 4;
  float4 acc = *(const float4*)(b_dec + d0);
  #pragma unroll
  for (int s = 0; s < K_TOP; ++s) {
    float val = sv[s];
    float4 w = *(const float4*)(Wt + (size_t)si[s] * D_DIM + d0);
    acc.x += val * w.x; acc.y += val * w.y; acc.z += val * w.z; acc.w += val * w.w;
  }
  *(float4*)(out + (size_t)b * D_DIM + d0) = acc;
}

// ----------------------------------------------------------------
extern "C" void kernel_launch(void* const* d_in, const int* in_sizes, int n_in,
                              void* d_out, int out_size, void* d_ws, size_t ws_size,
                              hipStream_t stream) {
  const float* x     = (const float*)d_in[0];
  const float* W_enc = (const float*)d_in[1];
  const float* b_enc = (const float*)d_in[2];
  const float* W_dec = (const float*)d_in[3];
  const float* b_dec = (const float*)d_in[4];

  float* out = (float*)d_out;                        // [B, D]
  float* h   = out + (size_t)B_ROWS * D_DIM;         // [B, H] (h_sparse region, used dense then masked)

  char* ws = (char*)d_ws;
  bf16_t* xb = (bf16_t*)ws;  ws += (size_t)B_ROWS * D_DIM * sizeof(bf16_t);
  bf16_t* wb = (bf16_t*)ws;  ws += (size_t)H_DIM  * D_DIM * sizeof(bf16_t);
  float*  Wt = (float*)ws;   ws += (size_t)H_DIM  * D_DIM * sizeof(float);
  float*  tv = (float*)ws;   ws += (size_t)B_ROWS * K_TOP * sizeof(float);
  int*   tix = (int*)ws;

  cvt_f32_bf16<<<(B_ROWS * D_DIM / 8 + 255) / 256, 256, 0, stream>>>(x, xb, B_ROWS * D_DIM);
  cvt_f32_bf16<<<(H_DIM  * D_DIM / 8 + 255) / 256, 256, 0, stream>>>(W_enc, wb, H_DIM * D_DIM);
  transpose_wdec<<<dim3(H_DIM / 32, D_DIM / 32), dim3(32, 8), 0, stream>>>(W_dec, Wt);
  sae_encoder_gemm<<<dim3(B_ROWS / BM, H_DIM / BN), 256, 0, stream>>>(xb, wb, b_enc, h);
  sae_topk<<<B_ROWS, 256, 0, stream>>>(h, tv, tix);
  sae_decoder<<<B_ROWS, 256, 0, stream>>>(tv, tix, Wt, b_dec, out);
}
